// Deter_738734375713
// MI455X (gfx1250) — matrix operand order repaired and checked by reading, not compiled, not run on hardware
//
#include <hip/hip_runtime.h>
#include <hip/hip_bf16.h>

// ---------------------------------------------------------------------------
// Types for CDNA5 WMMA (wave32): D(f32 16x16) = A(bf16 16x32) x B(bf16 32x16) + C
// ---------------------------------------------------------------------------
typedef __attribute__((ext_vector_type(16))) __bf16 bf16x16;
typedef __attribute__((ext_vector_type(8)))  float  f32x8;
typedef __attribute__((ext_vector_type(4)))  unsigned int u32x4;
typedef __attribute__((ext_vector_type(8)))  int i32x8;
typedef __attribute__((ext_vector_type(4)))  int i32x4;

#if __has_builtin(__builtin_amdgcn_tensor_load_to_lds)
#define HAVE_TDM 1
#else
#define HAVE_TDM 0
#endif

union Frag {                 // 32 bytes: two 16B LDS reads -> one WMMA operand
    struct { uint4 lo, hi; } q;
    bf16x16 v;
};

__device__ __forceinline__ unsigned short f32_to_bf16(float f) {
    unsigned int u = __float_as_uint(f);
    u += 0x7FFFu + ((u >> 16) & 1u);          // round-to-nearest-even
    return (unsigned short)(u >> 16);
}

// ---------------------------------------------------------------------------
// fp32 -> bf16 conversion (4-wide), and action normalization + conversion
// ---------------------------------------------------------------------------
__global__ void __launch_bounds__(256) cvt4_kernel(const float* __restrict__ in,
                                                   unsigned short* __restrict__ out,
                                                   int n4) {
    int i = blockIdx.x * 256 + threadIdx.x;
    if (i < n4) {
        float4 v = reinterpret_cast<const float4*>(in)[i];
        ushort4 r;
        r.x = f32_to_bf16(v.x); r.y = f32_to_bf16(v.y);
        r.z = f32_to_bf16(v.z); r.w = f32_to_bf16(v.w);
        reinterpret_cast<ushort4*>(out)[i] = r;
    }
}

__global__ void __launch_bounds__(256) act_prep_kernel(const float* __restrict__ in,
                                                       unsigned short* __restrict__ out,
                                                       int n4) {
    int i = blockIdx.x * 256 + threadIdx.x;
    if (i < n4) {
        float4 v = reinterpret_cast<const float4*>(in)[i];
        ushort4 r;
        r.x = f32_to_bf16(v.x / fmaxf(fabsf(v.x), 1.f));
        r.y = f32_to_bf16(v.y / fmaxf(fabsf(v.y), 1.f));
        r.z = f32_to_bf16(v.z / fmaxf(fabsf(v.z), 1.f));
        r.w = f32_to_bf16(v.w / fmaxf(fabsf(v.w), 1.f));
        reinterpret_cast<ushort4*>(out)[i] = r;
    }
}

// ---------------------------------------------------------------------------
// RMSNorm(+gain) + SiLU over `cols` elements of each row, f32 in -> bf16 out
// ---------------------------------------------------------------------------
__global__ void __launch_bounds__(256) norm_silu_kernel(
    const float* __restrict__ Y, int ldy, int coloff, int cols,
    const float* __restrict__ gain,
    unsigned short* __restrict__ Out, int ldo, int ocoloff) {
    __shared__ float red[256];
    const int row = blockIdx.x, tid = threadIdx.x;
    const float* y = Y + (size_t)row * ldy + coloff;
    float ss = 0.f;
    for (int c = tid; c < cols; c += 256) { float v = y[c]; ss += v * v; }
    red[tid] = ss;
    __syncthreads();
    for (int s = 128; s > 0; s >>= 1) {
        if (tid < s) red[tid] += red[tid + s];
        __syncthreads();
    }
    const float scale = rsqrtf(red[0] / (float)cols + 1e-4f);
    unsigned short* o = Out + (size_t)row * ldo + ocoloff;
    for (int c = tid; c < cols; c += 256) {
        float t = y[c] * scale * gain[c];
        float s = t / (1.f + __expf(-t));     // SiLU
        o[c] = f32_to_bf16(s);
    }
}

// ---------------------------------------------------------------------------
// Tiled bf16 WMMA GEMM.
//   C[M,N] (per block g) = concat(A0[:, g*a0_gstride : +acols0], A1) @ B_g + bias
// Tile: BM=64, BN=128, BK=32. 256 threads = 8 waves, wave grid 4(M) x 2(N),
// wave tile 16 x 64 (four 16x16x32 WMMAs per k-step per chunk).
// A tile staged via the Tensor Data Mover (TDM padding produces the LDK=40
// layout directly); B staged K-transposed via ds stores so fragments are
// contiguous ds_load_b128 pairs.
// NCH=1: plain f32 store epilogue.  NCH=3 + GRU: fused reset/cand/update GRU.
// ---------------------------------------------------------------------------
#define BM 64
#define BN 128
#define BK 32
#define NT 4     // 16-col WMMA tiles per wave
#define LDK 40   // padded K stride in LDS (halves): 32 data + 8 pad

template <int NCH, bool GRU>
__global__ void __launch_bounds__(256) gemm_bf16_kernel(
    const unsigned short* __restrict__ A0, int lda0, int acols0, int a0_gstride,
    const unsigned short* __restrict__ A1, int lda1,
    const unsigned short* __restrict__ Bw, int ldb, long long b_gstride, int chunk_stride,
    const float* __restrict__ bias, int bias_gstride,
    float* __restrict__ Cout, int ldc, int c_coloff, int c_gstride,
    int Ktot,
    const float* __restrict__ deter, float* __restrict__ out) {

    __shared__ unsigned short ldsA[BM * LDK];
    __shared__ unsigned short ldsB[NCH * BN * LDK];

    const int tid = threadIdx.x;
    const int g  = blockIdx.z;
    const int m0 = blockIdx.y * BM;
    const int n0 = blockIdx.x * BN;

    const int a0off = g * a0_gstride;
    const unsigned short* Bg = Bw + (long long)g * b_gstride;

    // loader roles
    const int bk  = tid >> 3;            // B k-row 0..31
    const int bcg = (tid & 7) * 16;      // B col-seg (16 halves)

    // wave / lane geometry
    const int wave  = tid >> 5;
    const int lane  = tid & 31;
    const int laneh = lane >> 4;         // 0/1
    const int lr    = lane & 15;
    const int mw    = wave >> 1;         // 0..3
    const int nw    = wave & 1;          // 0..1
    const int arow  = mw * 16 + lr;
    const int klo   = laneh * 8;         // A frag K base within 0..15
    const int khalf = laneh * 8;         // B frag K base within 0..15

    f32x8 acc[NCH][NT];
#pragma unroll
    for (int c = 0; c < NCH; ++c)
#pragma unroll
        for (int nt = 0; nt < NT; ++nt)
            acc[c][nt] = f32x8{0.f, 0.f, 0.f, 0.f, 0.f, 0.f, 0.f, 0.f};

    for (int k0 = 0; k0 < Ktot; k0 += BK) {
        // ---- B global loads into registers (2x16B per chunk) ----
        uint4 bv0[NCH], bv1[NCH];
#pragma unroll
        for (int c = 0; c < NCH; ++c) {
            const unsigned short* bp =
                Bg + (size_t)(k0 + bk) * ldb + c * chunk_stride + n0 + bcg;
            bv0[c] = *reinterpret_cast<const uint4*>(bp);
            bv1[c] = *reinterpret_cast<const uint4*>(bp + 8);
        }
        if (k0 + BK < Ktot)   // hint next B k-tile (global_prefetch_b8)
            __builtin_prefetch(Bg + (size_t)(k0 + BK + bk) * ldb + n0 + bcg, 0, 1);

#if !HAVE_TDM
        uint4 av;
        {
            const int ar  = tid >> 2;        // A row 0..63
            const int asg = (tid & 3) * 8;   // A k-seg (8 halves)
            const int k = k0 + asg;
            const unsigned short* ap =
                (k < acols0) ? (A0 + (size_t)(m0 + ar) * lda0 + a0off + k)
                             : (A1 + (size_t)(m0 + ar) * lda1 + (k - acols0));
            av = *reinterpret_cast<const uint4*>(ap);
        }
#endif

        __syncthreads();      // previous iteration's LDS reads complete

        // ---- stage B transposed to [col][k] ----
#pragma unroll
        for (int c = 0; c < NCH; ++c) {
            const unsigned short* p0 = reinterpret_cast<const unsigned short*>(&bv0[c]);
            const unsigned short* p1 = reinterpret_cast<const unsigned short*>(&bv1[c]);
#pragma unroll
            for (int i = 0; i < 8; ++i) {
                ldsB[(c * BN + bcg + i) * LDK + bk]     = p0[i];
                ldsB[(c * BN + bcg + 8 + i) * LDK + bk] = p1[i];
            }
        }

        // ---- stage A tile [64 x 32] -> ldsA (row stride LDK) ----
#if HAVE_TDM
        if (tid < 32) {                      // TDM ignores EXEC: one wave only
            const unsigned short* asrc; int alda, acol;
            if (k0 < acols0) { asrc = A0; alda = lda0; acol = a0off + k0; }
            else             { asrc = A1; alda = lda1; acol = k0 - acols0; }
            const unsigned long long ga =
                (unsigned long long)(uintptr_t)(asrc + (size_t)m0 * alda + acol);
            const unsigned int ldsa = (unsigned int)(uintptr_t)(&ldsA[0]);

            // D# group 0: count=1 | lds_addr | 57b global_addr | type=2
            u32x4 g0;
            g0[0] = 1u;
            g0[1] = ldsa;
            g0[2] = (unsigned int)(ga & 0xFFFFFFFFull);
            g0[3] = (unsigned int)((ga >> 32) & 0x01FFFFFFull) | (2u << 30);

            // D# group 1: data_size=2B, pad_enable, pad_interval=16 DW (enc 3),
            // pad_amount=4 DW (enc 3) -> 32 halves data + 8 halves pad = LDK.
            // tensor dims set to 2^30 (always in-bounds), tile 32 x 64,
            // tensor_dim0_stride = lda (elements).
            i32x8 g1;
            g1[0] = (1 << 16) | (1 << 20) | (3 << 22) | (3 << 25);
            g1[1] = 0;                                   // dim0 low16 (2^30 -> 0)
            g1[2] = (int)(0x4000u);                      // dim0 hi16; dim1 low16=0
            g1[3] = (int)(0x4000u) | (BK << 16);         // dim1 hi16 | tile_dim0=32
            g1[4] = BM;                                  // tile_dim1=64, tile_dim2=0
            g1[5] = alda;                                // dim0_stride[31:0]
            g1[6] = 0;                                   // dim0_stride[47:32]
            g1[7] = 0;
            i32x4 gz = {0, 0, 0, 0};
#if defined(__clang_major__) && __clang_major__ >= 23
            i32x8 gz8 = {0, 0, 0, 0, 0, 0, 0, 0};
            __builtin_amdgcn_tensor_load_to_lds(g0, g1, gz, gz, gz8, 0);
#else
            __builtin_amdgcn_tensor_load_to_lds(g0, g1, gz, gz, 0);
#endif
#if __has_builtin(__builtin_amdgcn_s_wait_tensorcnt)
            __builtin_amdgcn_s_wait_tensorcnt(0);
#else
            asm volatile("s_wait_tensorcnt 0x0" ::: "memory");
#endif
        }
#else
        {
            const int ar  = tid >> 2;
            const int asg = (tid & 3) * 8;
            *reinterpret_cast<uint4*>(&ldsA[ar * LDK + asg]) = av;
        }
#endif
        __syncthreads();

        // ---- WMMA compute ----
        Frag a;
        a.q.lo = *reinterpret_cast<const uint4*>(&ldsA[arow * LDK + klo]);
        a.q.hi = *reinterpret_cast<const uint4*>(&ldsA[arow * LDK + klo + 16]);
#pragma unroll
        for (int c = 0; c < NCH; ++c) {
#pragma unroll
            for (int nt = 0; nt < NT; ++nt) {
                const int col = nw * 64 + nt * 16 + lr;
                const unsigned short* bp = &ldsB[(c * BN + col) * LDK + khalf];
                Frag b;
                b.q.lo = *reinterpret_cast<const uint4*>(bp);
                b.q.hi = *reinterpret_cast<const uint4*>(bp + 16);
                acc[c][nt] = __builtin_amdgcn_wmma_f32_16x16x32_bf16(
                    false, a.v, false, b.v, (short)0, acc[c][nt], false, false);
            }
        }
    }

    // ---- epilogue ----
    const int rowb = m0 + mw * 16 + laneh * 8;     // C layout: VGPR r -> M = r (+8 hi lanes)
    if (!GRU) {
        const float* bp = bias + (size_t)g * bias_gstride;
        float* cp = Cout + c_coloff + (size_t)g * c_gstride;
#pragma unroll
        for (int nt = 0; nt < NT; ++nt) {
            const int col = n0 + nw * 64 + nt * 16 + lr;
            const float bvv = bp[col];
#pragma unroll
            for (int r = 0; r < 8; ++r)
                cp[(size_t)(rowb + r) * ldc + col] = acc[0][nt][r] + bvv;
        }
    } else {
        const float* gbp = bias + (size_t)g * bias_gstride;   // [reset|cand|update] x 512
#pragma unroll
        for (int nt = 0; nt < NT; ++nt) {
            const int col = n0 + nw * 64 + nt * 16 + lr;      // within block's 512
            const float br = gbp[col], bc = gbp[512 + col], bu = gbp[1024 + col];
#pragma unroll
            for (int r = 0; r < 8; ++r) {
                const int row = rowb + r;
                const float reset = 1.f / (1.f + __expf(-(acc[0][nt][r] + br)));
                const float cand  = tanhf(reset * (acc[1][nt][r] + bc));
                const float upd   = 1.f / (1.f + __expf(-((acc[2][nt][r] + bu) - 1.f)));
                const size_t oi = (size_t)row * 4096 + (size_t)g * 512 + col;
                out[oi] = upd * cand + (1.f - upd) * deter[oi];
            }
        }
    }
}

// ---------------------------------------------------------------------------
// Host launch
// ---------------------------------------------------------------------------
extern "C" void kernel_launch(void* const* d_in, const int* in_sizes, int n_in,
                              void* d_out, int out_size, void* d_ws, size_t ws_size,
                              hipStream_t stream) {
    (void)in_sizes; (void)n_in; (void)out_size; (void)ws_size;

    const float* stoch  = (const float*)d_in[0];   // [1024,1024]
    const float* deter  = (const float*)d_in[1];   // [1024,4096]
    const float* action = (const float*)d_in[2];   // [1024,128]
    const float* w0 = (const float*)d_in[3];       // [4096,1024]
    const float* b0 = (const float*)d_in[4];
    const float* g0 = (const float*)d_in[5];
    const float* w1 = (const float*)d_in[6];       // [1024,1024]
    const float* b1 = (const float*)d_in[7];
    const float* g1 = (const float*)d_in[8];
    const float* w2 = (const float*)d_in[9];       // [128,1024]
    const float* b2 = (const float*)d_in[10];
    const float* g2 = (const float*)d_in[11];
    const float* hw0 = (const float*)d_in[12];     // [8,3584,512]
    const float* hb0 = (const float*)d_in[13];
    const float* hg0 = (const float*)d_in[14];
    const float* hw1 = (const float*)d_in[15];     // [8,512,512]
    const float* hb1 = (const float*)d_in[16];
    const float* hg1 = (const float*)d_in[17];
    const float* gw  = (const float*)d_in[18];     // [8,512,1536]
    const float* gb  = (const float*)d_in[19];     // [12288]
    float* outF = (float*)d_out;                   // [1024,4096]

    // ---- workspace carve-up (bytes, 256-aligned) ----
    char* ws = (char*)d_ws;
    size_t off = 0;
    auto carve = [&](size_t bytes) -> void* {
        void* p = ws + off;
        off = (off + bytes + 255) & ~(size_t)255;
        return p;
    };
    typedef unsigned short u16;
    const size_t nW0 = 4096ull * 1024, nW1 = 1024ull * 1024, nW2 = 128ull * 1024;
    const size_t nHW0 = 8ull * 3584 * 512, nHW1 = 8ull * 512 * 512, nGW = 8ull * 512 * 1536;

    u16* W0b  = (u16*)carve(nW0  * 2);
    u16* W1b  = (u16*)carve(nW1  * 2);
    u16* W2b  = (u16*)carve(nW2  * 2);
    u16* HW0b = (u16*)carve(nHW0 * 2);
    u16* HW1b = (u16*)carve(nHW1 * 2);
    u16* GWb  = (u16*)carve(nGW  * 2);
    u16* Adet = (u16*)carve(1024ull * 4096 * 2);
    u16* Asto = (u16*)carve(1024ull * 1024 * 2);
    u16* Aact = (u16*)carve(1024ull * 128  * 2);
    float* Y  = (float*)carve(1024ull * 4096 * 4);   // reused f32 GEMM output
    u16* Xb   = (u16*)carve(1024ull * 3072 * 2);     // concat branch outputs (bf16)
    u16* H0b  = (u16*)carve(1024ull * 4096 * 2);
    u16* H1b  = (u16*)carve(1024ull * 4096 * 2);

    // ---- fp32 -> bf16 conversions ----
    auto cvt = [&](const float* src, u16* dst, size_t n) {
        int n4 = (int)(n / 4);
        cvt4_kernel<<<(n4 + 255) / 256, 256, 0, stream>>>(src, dst, n4);
    };
    cvt(w0, W0b, nW0);
    cvt(w1, W1b, nW1);
    cvt(w2, W2b, nW2);
    cvt(hw0, HW0b, nHW0);
    cvt(hw1, HW1b, nHW1);
    cvt(gw, GWb, nGW);
    cvt(deter, Adet, 1024ull * 4096);
    cvt(stoch, Asto, 1024ull * 1024);
    {
        int n4 = (1024 * 128) / 4;
        act_prep_kernel<<<(n4 + 255) / 256, 256, 0, stream>>>(action, Aact, n4);
    }

    // ---- input branches: Y[:,0:3072] = [deter@w0 | stoch@w1 | action@w2] + bias ----
    gemm_bf16_kernel<1, false><<<dim3(1024 / BN, 1024 / BM, 1), 256, 0, stream>>>(
        Adet, 4096, 4096, 0, Adet, 0, W0b, 1024, 0, 0, b0, 0, Y, 3072, 0, 0, 4096,
        nullptr, nullptr);
    gemm_bf16_kernel<1, false><<<dim3(1024 / BN, 1024 / BM, 1), 256, 0, stream>>>(
        Asto, 1024, 1024, 0, Asto, 0, W1b, 1024, 0, 0, b1, 0, Y, 3072, 1024, 0, 1024,
        nullptr, nullptr);
    gemm_bf16_kernel<1, false><<<dim3(1024 / BN, 1024 / BM, 1), 256, 0, stream>>>(
        Aact, 128, 128, 0, Aact, 0, W2b, 1024, 0, 0, b2, 0, Y, 3072, 2048, 0, 128,
        nullptr, nullptr);

    // ---- per-branch rmsnorm(1024) + silu -> Xb (bf16, [1024,3072]) ----
    norm_silu_kernel<<<1024, 256, 0, stream>>>(Y, 3072, 0,    1024, g0, Xb, 3072, 0);
    norm_silu_kernel<<<1024, 256, 0, stream>>>(Y, 3072, 1024, 1024, g1, Xb, 3072, 1024);
    norm_silu_kernel<<<1024, 256, 0, stream>>>(Y, 3072, 2048, 1024, g2, Xb, 3072, 2048);

    // ---- hidden0: per block, A = [deter_block(512) | Xb(3072)], K=3584, N=512 ----
    gemm_bf16_kernel<1, false><<<dim3(512 / BN, 1024 / BM, 8), 256, 0, stream>>>(
        Adet, 4096, 512, 512, Xb, 3072, HW0b, 512, 3584LL * 512, 0, hb0, 512,
        Y, 4096, 0, 512, 3584, nullptr, nullptr);
    norm_silu_kernel<<<1024, 256, 0, stream>>>(Y, 4096, 0, 4096, hg0, H0b, 4096, 0);

    // ---- hidden1: per block, K=512, N=512 ----
    gemm_bf16_kernel<1, false><<<dim3(512 / BN, 1024 / BM, 8), 256, 0, stream>>>(
        H0b, 4096, 512, 512, H0b, 0, HW1b, 512, 512LL * 512, 0, hb1, 512,
        Y, 4096, 0, 512, 512, nullptr, nullptr);
    norm_silu_kernel<<<1024, 256, 0, stream>>>(Y, 4096, 0, 4096, hg1, H1b, 4096, 0);

    // ---- gate GEMM (3 chunks fused) + GRU combine -> d_out ----
    gemm_bf16_kernel<3, true><<<dim3(512 / BN, 1024 / BM, 8), 256, 0, stream>>>(
        H1b, 4096, 512, 512, H1b, 0, GWb, 1536, 512LL * 1536, 512, gb, 1536,
        nullptr, 0, 0, 0, 512, deter, outF);
}
